// TransformerLMDecoder_5918464934499
// MI455X (gfx1250) — matrix operand order repaired and checked by reading, not compiled, not run on hardware
//
#include <hip/hip_runtime.h>
#include <hip/hip_bf16.h>

// ---------------- types for WMMA ----------------
typedef __bf16 bf16_t;
typedef __bf16 v16bf __attribute__((ext_vector_type(16)));
typedef __bf16 v8bf  __attribute__((ext_vector_type(8)));
typedef float  v8f   __attribute__((ext_vector_type(8)));

#define S_LEN 2048
#define BATCH 2
#define EMB   1024
#define HEADS 16
#define HDIM  64
#define NHEAD 32    // BATCH*HEADS
#define MROWS 4096  // S_LEN*BATCH

// ---- CDNA5 async global->LDS path (GLOBAL_LOAD_ASYNC_TO_LDS_B128, ASYNCcnt) ----
#if defined(__has_builtin)
#  if __has_builtin(__builtin_amdgcn_global_load_async_to_lds_b128)
#    define HAS_ASYNC_LDS 1
#  endif
#endif
#ifndef HAS_ASYNC_LDS
#  define HAS_ASYNC_LDS 0
#endif

// exact pointee type from the hipcc diagnostic: v4i in AS(1) / AS(3)
typedef int v4i_ __attribute__((vector_size(4 * sizeof(int))));
typedef __attribute__((address_space(1))) v4i_ g_v4i;
typedef __attribute__((address_space(3))) v4i_ l_v4i;

// copy 16 bytes global -> LDS (async when available)
static __device__ inline void cp16_g2l(const void* g, void* l) {
#if HAS_ASYNC_LDS
  __builtin_amdgcn_global_load_async_to_lds_b128((g_v4i*)g, (l_v4i*)l, 0, 0);
#else
  *(uint4*)l = *(const uint4*)g;
#endif
}

static __device__ inline void wait_async_all() {
#if HAS_ASYNC_LDS
#  if __has_builtin(__builtin_amdgcn_s_wait_asynccnt)
  __builtin_amdgcn_s_wait_asynccnt(0);
#  else
  asm volatile("s_wait_asynccnt 0x0" ::: "memory");
#  endif
#endif
}

// round-to-nearest-even f32 -> bf16 (pure integer path)
static __device__ inline bf16_t f2bf(float f) {
  unsigned u = __builtin_bit_cast(unsigned, f);
  unsigned r = u + 0x7fffu + ((u >> 16) & 1u);
  unsigned short h = (unsigned short)(r >> 16);
  return __builtin_bit_cast(bf16_t, h);
}

// build a 16-element bf16 fragment from two 8-element (16B) runs
static __device__ inline v16bf ld16(const bf16_t* p0, const bf16_t* p1) {
  v8bf lo = *(const v8bf*)p0;
  v8bf hi = *(const v8bf*)p1;
  return __builtin_shufflevector(lo, hi, 0,1,2,3,4,5,6,7,8,9,10,11,12,13,14,15);
}

static __device__ inline v8f wmma_bf16(v16bf a, v16bf b, v8f c) {
  return __builtin_amdgcn_wmma_f32_16x16x32_bf16(false, a, false, b, (short)0, c, false, false);
}

// ---------------- fp32 -> bf16 convert ----------------
__global__ void f32_to_bf16_kernel(const float* __restrict__ src,
                                   bf16_t* __restrict__ dst, int n) {
  int i = blockIdx.x * blockDim.x + threadIdx.x;
  if (i < n) dst[i] = f2bf(src[i]);
}

// ---------------- QKV GEMM ----------------
// qkv[r, f] = (sum_k x[r,k] * W[f,k] + bias[f]) * gamma[f]   (q additionally / sqrt(D))
// Block tile: 128 (M) x 64 (N), 8 waves, each wave 16x64 output. K step = 32.
// Double-buffered LDS tiles filled by async global->LDS copies; 1 barrier/step.
__global__ __launch_bounds__(256)
void qkv_gemm_kernel(const bf16_t* __restrict__ X, const bf16_t* __restrict__ W,
                     const float* __restrict__ bias, const float* __restrict__ gamma,
                     bf16_t* __restrict__ q_ws, bf16_t* __restrict__ k_ws,
                     bf16_t* __restrict__ vT_ws) {
  __shared__ bf16_t As[2][128 * 32];
  __shared__ bf16_t Bs[2][64 * 32];
  const int tid  = threadIdx.x;
  const int wave = tid >> 5, lane = tid & 31;
  const int half = lane >> 4, nl = lane & 15;
  const int m_base = blockIdx.x * 128;
  const int n_base = blockIdx.y * 64;

  v8f acc[4];
#pragma unroll
  for (int c = 0; c < 4; ++c)
#pragma unroll
    for (int i = 0; i < 8; ++i) acc[c][i] = 0.0f;

  const int arow = tid >> 1, akp = (tid & 1) * 16;  // 128x32: 16 elems/thread
  const int brow = tid >> 2, bkp = (tid & 3) * 8;   // 64x32:  8 elems/thread
  const bf16_t* gA = X + (size_t)(m_base + arow) * EMB + akp;
  const bf16_t* gB = W + (size_t)(n_base + brow) * EMB + bkp;

  auto copy_tiles = [&](int buf, int k0) {
    cp16_g2l(gA + k0,     &As[buf][arow * 32 + akp]);
    cp16_g2l(gA + k0 + 8, &As[buf][arow * 32 + akp + 8]);
    cp16_g2l(gB + k0,     &Bs[buf][brow * 32 + bkp]);
  };

  copy_tiles(0, 0);
  for (int k0 = 0; k0 < EMB; k0 += 32) {
    const int cur = (k0 >> 5) & 1;
    wait_async_all();
    __syncthreads();            // tile[cur] ready; tile[cur^1] reads from prev iter done
    if (k0 + 32 < EMB) copy_tiles(cur ^ 1, k0 + 32);

    // A fragment: row m = wave*16 + nl, K runs at half*16 .. half*16+15
    const bf16_t* ap = &As[cur][(wave * 16 + nl) * 32];
    v16bf af = ld16(ap + half * 16, ap + half * 16 + 8);
    // preload all 4 B fragments, then back-to-back WMMA
    v16bf bfr[4];
#pragma unroll
    for (int c = 0; c < 4; ++c) {
      const bf16_t* bp = &Bs[cur][(c * 16 + nl) * 32 + half * 16];
      bfr[c] = ld16(bp, bp + 8);
    }
#pragma unroll
    for (int c = 0; c < 4; ++c) acc[c] = wmma_bf16(af, bfr[c], acc[c]);
  }

  // epilogue: C layout -> scatter. lane: col = c*16+nl, rows = half*8 + i
  const int rbase = m_base + wave * 16 + half * 8;
#pragma unroll
  for (int c = 0; c < 4; ++c) {
    const int f = n_base + c * 16 + nl;
    const int which = f >> 10;         // 0=q 1=k 2=v (uniform across the 16-col tile)
    const int h = (f & 1023) >> 6;
    const int d = f & 63;
    const float bsc = bias[f];
    float gsc = gamma[f];
    if (which == 0) gsc *= 0.125f;     // fold 1/sqrt(64) into q
#pragma unroll
    for (int i = 0; i < 8; ++i) {
      const int r = rbase + i;
      const int s = r >> 1, b = r & 1;
      const int head = b * HEADS + h;
      bf16_t bv = f2bf((acc[c][i] + bsc) * gsc);
      if (which == 0)      q_ws[((size_t)head * S_LEN + s) * HDIM + d] = bv;
      else if (which == 1) k_ws[((size_t)head * S_LEN + s) * HDIM + d] = bv;
      else                 vT_ws[((size_t)head * HDIM + d) * S_LEN + s] = bv;
    }
  }
}

// ---------------- causal flash attention ----------------
// Block: 128 threads = 4 waves; wave owns 16 query rows of one head.
// Per 32-key tile: 4 WMMA for scores (D=64 as 2x K=32), online softmax with
// width-16 shuffle row reductions, P transposed C->A layout via per-wave LDS,
// 4 WMMA for P @ V (vT gives contiguous B fragments).
__global__ __launch_bounds__(128)
void flash_attn_kernel(const bf16_t* __restrict__ q_ws, const bf16_t* __restrict__ k_ws,
                       const bf16_t* __restrict__ vT_ws, const float* __restrict__ lin_alpha,
                       bf16_t* __restrict__ attn_ws) {
  __shared__ bf16_t Ps[4 * 16 * 32];  // per-wave 16x32 probability tile
  const int tid  = threadIdx.x;
  const int wave = tid >> 5, lane = tid & 31;
  const int half = lane >> 4, nl = lane & 15;
  const int head  = blockIdx.y;
  const int qbase = blockIdx.x * 64 + wave * 16;
  const int b_idx = head >> 4, hh = head & 15;

  // Q A-fragments (16x32 each): rows = queries, K = d
  const bf16_t* qrow = q_ws + ((size_t)head * S_LEN + qbase + nl) * HDIM;
  v16bf qa0 = ld16(qrow + half * 16,      qrow + half * 16 + 8);      // d 0..31
  v16bf qa1 = ld16(qrow + 32 + half * 16, qrow + 40 + half * 16);     // d 32..63

  v8f acc[4];
#pragma unroll
  for (int c = 0; c < 4; ++c)
#pragma unroll
    for (int i = 0; i < 8; ++i) acc[c][i] = 0.0f;
  float mrow[8], lrow[8];
#pragma unroll
  for (int i = 0; i < 8; ++i) { mrow[i] = -1e30f; lrow[i] = 0.0f; }

  bf16_t* pbuf = Ps + wave * 512;
  const int qend = qbase + 15;

  for (int kt = 0; kt <= qend; kt += 32) {
    v8f s0, s1;
#pragma unroll
    for (int i = 0; i < 8; ++i) { s0[i] = 0.0f; s1[i] = 0.0f; }

    // K^T B-fragments: col = key, K-dim = d (contiguous per key row)
    const bf16_t* kr0 = k_ws + ((size_t)head * S_LEN + kt + nl) * HDIM;
    const bf16_t* kr1 = kr0 + 16 * HDIM;
    v16bf kb00 = ld16(kr0 + half * 16,      kr0 + half * 16 + 8);     // keys kt..+15, d 0..31
    v16bf kb01 = ld16(kr0 + 32 + half * 16, kr0 + 40 + half * 16);    // d 32..63
    v16bf kb10 = ld16(kr1 + half * 16,      kr1 + half * 16 + 8);     // keys kt+16..+31
    v16bf kb11 = ld16(kr1 + 32 + half * 16, kr1 + 40 + half * 16);

    s0 = wmma_bf16(qa0, kb00, s0);  s0 = wmma_bf16(qa1, kb01, s0);
    s1 = wmma_bf16(qa0, kb10, s1);  s1 = wmma_bf16(qa1, kb11, s1);

    // causal mask (C layout: row = qbase + half*8 + i, col = kt(+16) + nl)
#pragma unroll
    for (int i = 0; i < 8; ++i) {
      const int srow = qbase + half * 8 + i;
      if (kt + nl > srow)      s0[i] = -1e30f;
      if (kt + 16 + nl > srow) s1[i] = -1e30f;
    }

    float alf[8];
#pragma unroll
    for (int i = 0; i < 8; ++i) {
      float t = fmaxf(s0[i], s1[i]);
      t = fmaxf(t, __shfl_xor(t, 1, 16));
      t = fmaxf(t, __shfl_xor(t, 2, 16));
      t = fmaxf(t, __shfl_xor(t, 4, 16));
      t = fmaxf(t, __shfl_xor(t, 8, 16));
      const float mnew = fmaxf(mrow[i], t);
      const float al = __expf(mrow[i] - mnew);
      const float p0 = __expf(s0[i] - mnew);
      const float p1 = __expf(s1[i] - mnew);
      float rs = p0 + p1;
      rs += __shfl_xor(rs, 1, 16);
      rs += __shfl_xor(rs, 2, 16);
      rs += __shfl_xor(rs, 4, 16);
      rs += __shfl_xor(rs, 8, 16);
      lrow[i] = lrow[i] * al + rs;
      mrow[i] = mnew;
      alf[i] = al;
      pbuf[(half * 8 + i) * 32 + nl]      = f2bf(p0);   // C->A transpose via LDS
      pbuf[(half * 8 + i) * 32 + 16 + nl] = f2bf(p1);
    }
#pragma unroll
    for (int c = 0; c < 4; ++c)
#pragma unroll
      for (int i = 0; i < 8; ++i) acc[c][i] *= alf[i];

    // P as A-fragment (rows = queries, K = 32 keys); same-wave LDS is in-order
    const bf16_t* prow = pbuf + nl * 32;
    v16bf pa = ld16(prow + half * 16, prow + half * 16 + 8);
    // preload V B-fragments from vT, then back-to-back WMMA
    v16bf vb[4];
#pragma unroll
    for (int c = 0; c < 4; ++c) {
      const bf16_t* vrow = vT_ws + ((size_t)head * HDIM + c * 16 + nl) * S_LEN + kt + half * 16;
      vb[c] = ld16(vrow, vrow + 8);
    }
#pragma unroll
    for (int c = 0; c < 4; ++c) acc[c] = wmma_bf16(pa, vb[c], acc[c]);
  }

  // epilogue: out = (acc/l) * linear_alpha[e], bf16 to attn_ws[r, e]
#pragma unroll
  for (int c = 0; c < 4; ++c) {
    const int e = hh * HDIM + c * 16 + nl;
    const float la = lin_alpha[e];
#pragma unroll
    for (int i = 0; i < 8; ++i) {
      const int srow = qbase + half * 8 + i;
      const int r = srow * BATCH + b_idx;
      attn_ws[(size_t)r * EMB + e] = f2bf(acc[c][i] / lrow[i] * la);
    }
  }
}

// ---------------- out projection ----------------
// out[r,f] = (sum_k attn[r,k]*W[f,k] + bias[f]) * lin_gamma[f]  -> fp32
__global__ __launch_bounds__(256)
void out_proj_gemm_kernel(const bf16_t* __restrict__ Xa, const bf16_t* __restrict__ W,
                          const float* __restrict__ bias, const float* __restrict__ lgamma,
                          float* __restrict__ out) {
  __shared__ bf16_t As[2][128 * 32];
  __shared__ bf16_t Bs[2][64 * 32];
  const int tid  = threadIdx.x;
  const int wave = tid >> 5, lane = tid & 31;
  const int half = lane >> 4, nl = lane & 15;
  const int m_base = blockIdx.x * 128;
  const int n_base = blockIdx.y * 64;

  v8f acc[4];
#pragma unroll
  for (int c = 0; c < 4; ++c)
#pragma unroll
    for (int i = 0; i < 8; ++i) acc[c][i] = 0.0f;

  const int arow = tid >> 1, akp = (tid & 1) * 16;
  const int brow = tid >> 2, bkp = (tid & 3) * 8;
  const bf16_t* gA = Xa + (size_t)(m_base + arow) * EMB + akp;
  const bf16_t* gB = W + (size_t)(n_base + brow) * EMB + bkp;

  auto copy_tiles = [&](int buf, int k0) {
    cp16_g2l(gA + k0,     &As[buf][arow * 32 + akp]);
    cp16_g2l(gA + k0 + 8, &As[buf][arow * 32 + akp + 8]);
    cp16_g2l(gB + k0,     &Bs[buf][brow * 32 + bkp]);
  };

  copy_tiles(0, 0);
  for (int k0 = 0; k0 < EMB; k0 += 32) {
    const int cur = (k0 >> 5) & 1;
    wait_async_all();
    __syncthreads();
    if (k0 + 32 < EMB) copy_tiles(cur ^ 1, k0 + 32);

    const bf16_t* ap = &As[cur][(wave * 16 + nl) * 32];
    v16bf af = ld16(ap + half * 16, ap + half * 16 + 8);
    v16bf bfr[4];
#pragma unroll
    for (int c = 0; c < 4; ++c) {
      const bf16_t* bp = &Bs[cur][(c * 16 + nl) * 32 + half * 16];
      bfr[c] = ld16(bp, bp + 8);
    }
#pragma unroll
    for (int c = 0; c < 4; ++c) acc[c] = wmma_bf16(af, bfr[c], acc[c]);
  }

  const int rbase = m_base + wave * 16 + half * 8;
#pragma unroll
  for (int c = 0; c < 4; ++c) {
    const int f = n_base + c * 16 + nl;
    const float bsc = bias[f];
    const float gsc = lgamma[f];
#pragma unroll
    for (int i = 0; i < 8; ++i) {
      const int r = rbase + i;
      out[(size_t)r * EMB + f] = (acc[c][i] + bsc) * gsc;
    }
  }
}

// ---------------- launch ----------------
extern "C" void kernel_launch(void* const* d_in, const int* in_sizes, int n_in,
                              void* d_out, int out_size, void* d_ws, size_t ws_size,
                              hipStream_t stream) {
  const float* x         = (const float*)d_in[0];
  const float* w_in      = (const float*)d_in[1];
  const float* b_in      = (const float*)d_in[2];
  const float* gamma     = (const float*)d_in[3];
  const float* lin_alpha = (const float*)d_in[4];
  const float* lin_gamma = (const float*)d_in[5];
  const float* w_out     = (const float*)d_in[6];
  const float* b_out     = (const float*)d_in[7];
  float* out = (float*)d_out;

  char* ws = (char*)d_ws;
  const size_t MB = 1024u * 1024u;
  bf16_t* xb      = (bf16_t*)(ws);             // 8 MB (reused as attn output)
  bf16_t* w_in_b  = (bf16_t*)(ws + 8 * MB);    // 6 MB
  bf16_t* w_out_b = (bf16_t*)(ws + 14 * MB);   // 2 MB
  bf16_t* q_ws    = (bf16_t*)(ws + 16 * MB);   // 8 MB  [head][s][d]
  bf16_t* k_ws    = (bf16_t*)(ws + 24 * MB);   // 8 MB  [head][s][d]
  bf16_t* vT_ws   = (bf16_t*)(ws + 32 * MB);   // 8 MB  [head][d][s]

  const int n1 = MROWS * EMB;      // x
  const int n2 = 3 * EMB * EMB;    // in_proj_weight
  const int n3 = EMB * EMB;        // out_proj_weight
  f32_to_bf16_kernel<<<(n1 + 255) / 256, 256, 0, stream>>>(x, xb, n1);
  f32_to_bf16_kernel<<<(n2 + 255) / 256, 256, 0, stream>>>(w_in, w_in_b, n2);
  f32_to_bf16_kernel<<<(n3 + 255) / 256, 256, 0, stream>>>(w_out, w_out_b, n3);

  qkv_gemm_kernel<<<dim3(32, 48), 256, 0, stream>>>(xb, w_in_b, b_in, gamma,
                                                    q_ws, k_ws, vT_ws);
  flash_attn_kernel<<<dim3(32, 32), 128, 0, stream>>>(q_ws, k_ws, vT_ws, lin_alpha,
                                                      xb /* attn output reuses xb */);
  out_proj_gemm_kernel<<<dim3(32, 16), 256, 0, stream>>>(xb, w_out_b, b_out, lin_gamma, out);
}
